// SNNAutoencoder_12317966205491
// MI455X (gfx1250) — matrix operand order repaired and checked by reading, not compiled, not run on hardware
//
#include <hip/hip_runtime.h>
#include <stdint.h>

typedef _Float16 v16h __attribute__((ext_vector_type(16)));
typedef _Float16 h8   __attribute__((ext_vector_type(8)));
typedef float    v8f  __attribute__((ext_vector_type(8)));
typedef float    f4v  __attribute__((ext_vector_type(4)));
typedef uint32_t u32;
typedef uint64_t u64;
typedef uint32_t u32x4 __attribute__((ext_vector_type(4)));
typedef uint32_t u32x8 __attribute__((ext_vector_type(8)));

#define D_IN  784
#define H_DIM 128
#define L_DIM 32
#define T_STEPS 16

// ---------------- LDS arena (66.5 KB, phase-overlapped) ----------------
// [     0 ..  8192) w2s : enc_w2 f16  [32][128]          (persistent)
// [  8192 .. 16384) w3s : dec_w1 f16  [128][32]          (persistent)
// [ 16384 .. 18432) h1b : h1 spike bits u32[8][16][4]    (phase 2)
// [ 18432 .. 18944) zbit: z  spike bits u32[8][16]       (phase 2)
// [ 18944 .. 51712) sb  : d1 spike bits u32[16][128][4]  (phase 2/3)
//                         also TDM f32 scratch (32 KB) before phase 1
// [ 51712 .. 68096) slot: enc_w1 K-chunk f16[2][128][32] dbl-buf (phase 1)
//                         dec_w2 N-tile f16[2][16][128] dbl-buf (phase 3)
#define W2S_OFF   0
#define W3S_OFF   8192
#define H1B_OFF   16384
#define ZBIT_OFF  18432
#define SB_OFF    18944
#define SLOT_OFF  51712
#define SMEM_SIZE 68096

static __device__ __forceinline__ v8f wmma_f16(v16h a, v16h b, v8f c) {
  // D = A(16x32 f16) * B(32x16 f16) + C(16x16 f32)
  return __builtin_amdgcn_wmma_f32_16x16x32_f16(false, a, false, b, (short)0, c,
                                                false, false);
}

// sigmoid via gfx1250 hardware v_tanh_f32 when available (1 trans op vs 2).
static __device__ __forceinline__ float fast_sigmoid(float v) {
#if __has_builtin(__builtin_amdgcn_tanhf)
  return 0.5f + 0.5f * __builtin_amdgcn_tanhf(0.5f * v);
#elif __has_builtin(__builtin_amdgcn_tanh_f32)
  return 0.5f + 0.5f * __builtin_amdgcn_tanh_f32(0.5f * v);
#else
  return 1.0f / (1.0f + __expf(-v));
#endif
}

// ---- TDM: 1-D tensor load global->LDS, n f32 elements (wave-uniform). ----
// D# group0: count=1 | lds_addr | global_addr[56:0] | type=2.
// D# group1: data_size=4B, tensor_dim0=n, tile_dim0=n, stride0=n, rest 0.
static __device__ __forceinline__ void tdm_load_lds_f32(u32 lds_off,
                                                        const float* gsrc,
                                                        u32 n) {
  u64 ga = (u64)(uintptr_t)gsrc;
  u32x4 g0;
  g0[0] = 1u;                                  // count=1, user mode, no gather
  g0[1] = lds_off;                             // lds_addr (bytes)
  g0[2] = (u32)ga;                             // global_addr[31:0]
  g0[3] = ((u32)(ga >> 32) & 0x01FFFFFFu) | 0x80000000u; // addr[56:32]|type=2
  u32x8 g1;
  g1[0] = 0x00020000u;                         // data_size=2 (4 bytes)
  g1[1] = (n & 0xFFFFu) << 16;                 // tensor_dim0[15:0]
  g1[2] = ((n >> 16) & 0xFFFFu) | (1u << 16);  // tensor_dim0[31:16]|td1=1
  g1[3] = (n & 0xFFFFu) << 16;                 // tile_dim0 = n
  g1[4] = 0u;                                  // tile_dim1=tile_dim2=0
  g1[5] = n;                                   // tensor_dim0_stride[31:0]
  g1[6] = 0u;
  g1[7] = 0u;
#pragma unroll
  for (int i = 0; i < 4; ++i)
    g0[i] = (u32)__builtin_amdgcn_readfirstlane((int)g0[i]);
#pragma unroll
  for (int i = 0; i < 8; ++i)
    g1[i] = (u32)__builtin_amdgcn_readfirstlane((int)g1[i]);
  asm volatile("tensor_load_to_lds %0, %1" : : "s"(g0), "s"(g1) : "memory");
}

// A-fragment (16x32 f16) of x rows from global f32 (converted).
// Lane: m = lane&15 (row), g = lane>>4. K(e) = g*16 + e.
static __device__ __forceinline__ v16h afrag_x(const float* __restrict__ row,
                                               int cb, int g) {
  v16h a;
  int c0 = cb + g * 16;
  int c1 = cb + g * 16 + 8;
  if (c0 + 8 <= D_IN) {
    f4v q0 = *(const f4v*)(row + c0);
    f4v q1 = *(const f4v*)(row + c0 + 4);
#pragma unroll
    for (int i = 0; i < 4; ++i) { a[i] = (_Float16)q0[i]; a[4 + i] = (_Float16)q1[i]; }
  } else {
#pragma unroll
    for (int i = 0; i < 8; ++i) a[i] = (_Float16)0.0f;
  }
  if (c1 + 8 <= D_IN) {
    f4v q0 = *(const f4v*)(row + c1);
    f4v q1 = *(const f4v*)(row + c1 + 4);
#pragma unroll
    for (int i = 0; i < 4; ++i) { a[8 + i] = (_Float16)q0[i]; a[12 + i] = (_Float16)q1[i]; }
  } else {
#pragma unroll
    for (int i = 0; i < 8; ++i) a[8 + i] = (_Float16)0.0f;
  }
  return a;
}

// B-fragment (32x16 f16) from 16 contiguous f16 in LDS (two ds_load_b128).
// Lane n = lane&15 (col), g = lane>>4, K(e) = g*16 + e.
static __device__ __forceinline__ v16h bfrag_lds(const _Float16* p) {
  h8 lo = *(const h8*)(p);
  h8 hi = *(const h8*)(p + 8);
  v16h b;
#pragma unroll
  for (int i = 0; i < 8; ++i) { b[i] = lo[i]; b[8 + i] = hi[i]; }
  return b;
}

// Expand 32 packed spike bits (one K=32 tile, row m=lane&15) into f16 A-frag.
static __device__ __forceinline__ v16h afrag_bits(u32 word, int g) {
  u32 b0 = (word >> (g * 16)) & 0xFFu;
  u32 b1 = (word >> (g * 16 + 8)) & 0xFFu;
  v16h a;
#pragma unroll
  for (int i = 0; i < 8; ++i) {
    a[i]     = ((b0 >> i) & 1u) ? (_Float16)1.0f : (_Float16)0.0f;
    a[8 + i] = ((b1 >> i) & 1u) ? (_Float16)1.0f : (_Float16)0.0f;
  }
  return a;
}

// LIF step (tau=2, vth=0.5, hard reset) on one 16x16 C-layout tile.
static __device__ __forceinline__ void lif_tile(v8f& v, const v8f& in, u32 bal[8]) {
#pragma unroll
  for (int r = 0; r < 8; ++r) {
    float nv = 0.5f * (v[r] + in[r]);
    bool sp = nv >= 0.5f;
    bal[r] = __builtin_amdgcn_ballot_w32(sp);
    v[r] = sp ? 0.0f : nv;
  }
}

// Pack two tile ballots (cols 0-15 / 16-31 of a 32-col group) into row words.
static __device__ __forceinline__ void store_bits(u32* base, int stride,
                                                  const u32 balE[8],
                                                  const u32 balO[8], bool lane0) {
  if (lane0) {
#pragma unroll
    for (int r = 0; r < 8; ++r) {
      base[r * stride]       = (balE[r] & 0xFFFFu) | (balO[r] << 16);
      base[(r + 8) * stride] = (balE[r] >> 16) | (balO[r] & 0xFFFF0000u);
    }
  }
}

__global__ __launch_bounds__(256) void snn_ae_fused(
    const float* __restrict__ x,
    const float* __restrict__ ew1, const float* __restrict__ eb1,
    const float* __restrict__ ew2, const float* __restrict__ eb2,
    const float* __restrict__ dw1, const float* __restrict__ db1,
    const float* __restrict__ dw2, const float* __restrict__ db2,
    float* __restrict__ out) {
  __shared__ __align__(16) unsigned char smem[SMEM_SIZE];
  _Float16* w2s  = (_Float16*)(smem + W2S_OFF);    // [32][128]
  _Float16* w3s  = (_Float16*)(smem + W3S_OFF);    // [128][32]
  u32* h1b  = (u32*)(smem + H1B_OFF);              // [8][16][4]
  u32* zbit = (u32*)(smem + ZBIT_OFF);             // [8][16]
  u32* sb   = (u32*)(smem + SB_OFF);               // [16][128][4]
  _Float16* slot = (_Float16*)(smem + SLOT_OFF);   // phase-overlapped, 16 KB

  const int tid  = threadIdx.x;
  const int lane = tid & 31;
  const int w    = tid >> 5;        // wave id 0..7
  const int g    = lane >> 4;
  const int n    = lane & 15;       // col-in-tile (B/C) == row-in-tile (A)
  const bool lane0 = (lane == 0);
  const int rowbase = blockIdx.x * 128 + w * 16;

  // ---- TDM-stage enc_w2 / dec_w1 (f32) into the sb region, then cvt f16 ----
  {
    float* scrA = (float*)(smem + SB_OFF);            // 16 KB
    float* scrB = (float*)(smem + SB_OFF + 16384);    // 16 KB
    if (w == 0) {
      u32 offA = (u32)(uintptr_t)scrA;
      u32 offB = (u32)(uintptr_t)scrB;
      tdm_load_lds_f32(offA, ew2, L_DIM * H_DIM);
      tdm_load_lds_f32(offB, dw1, H_DIM * L_DIM);
      __builtin_amdgcn_s_wait_tensorcnt(0);
    }
    __syncthreads();
#pragma unroll
    for (int i = tid; i < L_DIM * H_DIM; i += 256) {
      w2s[i] = (_Float16)scrA[i];
      w3s[i] = (_Float16)scrB[i];
    }
    __syncthreads();
  }

  // ---- Phase 1: x_in = x @ enc_w1^T + b1; enc_w1 K-chunks double-buffered --
  const float* xrow = x + (size_t)(rowbase + n) * D_IN;
  v8f xin[8];
#pragma unroll
  for (int ct = 0; ct < 8; ++ct) {
    float bv = eb1[ct * 16 + n];
#pragma unroll
    for (int r = 0; r < 8; ++r) xin[ct][r] = bv;
  }
  // prime buffer 0 with chunk kb=0 (cols 0..31)
  for (int i = tid; i < H_DIM * 32; i += 256) {
    int r = i >> 5, c = i & 31;
    slot[i] = (_Float16)ew1[(size_t)r * D_IN + c];
  }
  __syncthreads();
  for (int kb = 0; kb < 25; ++kb) {           // K = 784, zero-padded to 800
    int cb = kb * 32;
    if (kb + 1 < 25) {                        // stage next chunk -> other buf
      int cb2 = cb + 32;
      _Float16* dst = slot + ((kb + 1) & 1) * (H_DIM * 32);
      for (int i = tid; i < H_DIM * 32; i += 256) {
        int r = i >> 5, c = i & 31;
        int col = cb2 + c;
        dst[i] = (col < D_IN) ? (_Float16)ew1[(size_t)r * D_IN + col]
                              : (_Float16)0.0f;
      }
    }
    __builtin_prefetch(xrow + cb + 64, 0, 3);
    v16h a = afrag_x(xrow, cb, g);
    const _Float16* buf = slot + (kb & 1) * (H_DIM * 32);
    v16h bf[8];
#pragma unroll
    for (int ct = 0; ct < 8; ++ct)            // batch all ds loads first
      bf[ct] = bfrag_lds(&buf[(ct * 16 + n) * 32 + g * 16]);
#pragma unroll
    for (int ct = 0; ct < 8; ++ct)
      xin[ct] = wmma_f16(a, bf[ct], xin[ct]);
    __syncthreads();                          // stage/consume handshake
  }

  // ---- Phase 2: 16 LIF timesteps; spikes bit-packed into LDS ----
  v8f v1[8] = {}, v3[8] = {};
  v8f v2a = {}, v2b = {};
  const float eb2a = eb2[n], eb2b = eb2[16 + n];

  for (int t = 0; t < T_STEPS; ++t) {
    // LIF layer 1 -> h1 bits
#pragma unroll
    for (int cg = 0; cg < 4; ++cg) {
      u32 balE[8], balO[8];
      lif_tile(v1[2 * cg],     xin[2 * cg],     balE);
      lif_tile(v1[2 * cg + 1], xin[2 * cg + 1], balO);
      store_bits(&h1b[(w * 16) * 4 + cg], 4, balE, balO, lane0);
    }
    __syncthreads();

    // GEMM2: z_in = h1 @ enc_w2^T + b2   (16x128 @ 128x32)
    v8f za, zb2;
#pragma unroll
    for (int r = 0; r < 8; ++r) { za[r] = eb2a; zb2[r] = eb2b; }
#pragma unroll
    for (int kt = 0; kt < 4; ++kt) {
      v16h a = afrag_bits(h1b[(w * 16 + n) * 4 + kt], g);
      za  = wmma_f16(a, bfrag_lds(&w2s[n * H_DIM + kt * 32 + g * 16]),        za);
      zb2 = wmma_f16(a, bfrag_lds(&w2s[(16 + n) * H_DIM + kt * 32 + g * 16]), zb2);
    }
    // LIF layer 2 -> z bits
    {
      u32 balE[8], balO[8];
      lif_tile(v2a, za,  balE);
      lif_tile(v2b, zb2, balO);
      store_bits(&zbit[w * 16], 1, balE, balO, lane0);
    }
    __syncthreads();

    // GEMM3: d1_in = z @ dec_w1^T + b3   (16x32 @ 32x128), then LIF layer 3
    v16h az = afrag_bits(zbit[w * 16 + n], g);
#pragma unroll
    for (int cg = 0; cg < 4; ++cg) {
      float bE = db1[(2 * cg) * 16 + n];
      float bO = db1[(2 * cg + 1) * 16 + n];
      v8f dE, dO;
#pragma unroll
      for (int r = 0; r < 8; ++r) { dE[r] = bE; dO[r] = bO; }
      dE = wmma_f16(az, bfrag_lds(&w3s[((2 * cg) * 16 + n) * L_DIM + g * 16]),     dE);
      dO = wmma_f16(az, bfrag_lds(&w3s[((2 * cg + 1) * 16 + n) * L_DIM + g * 16]), dO);
      u32 balE[8], balO[8];
      lif_tile(v3[2 * cg],     dE, balE);
      lif_tile(v3[2 * cg + 1], dO, balO);
      store_bits(&sb[(t * 128 + w * 16) * 4 + cg], 4, balE, balO, lane0);
    }
    __syncthreads();   // WAR: h1b/zbit rewritten next step
  }

  // ---- Phase 3: acc = (1/16) * sum_t sigmoid(d1_t @ dec_w2^T + b4) ----
  // dec_w2 N-tiles cooperatively staged to LDS f16, double-buffered; the
  // 16-step sigmoid accumulation stays in VGPRs; out written exactly once.
  {
    // prime buffer 0 with tile nt=0
    for (int i = tid; i < 16 * H_DIM; i += 256) {
      int r = i >> 7, c = i & 127;
      slot[i] = (_Float16)dw2[(size_t)r * H_DIM + c];
    }
    __syncthreads();
  }
#pragma unroll 1
  for (int nt = 0; nt < 49; ++nt) {           // 784 / 16 column tiles
    int n0 = nt * 16;
    int buf = nt & 1;
    if (nt + 1 < 49) {                        // stage next tile into other buf
      const float* src = dw2 + (size_t)(n0 + 16) * H_DIM;
      if (nt + 2 < 49) __builtin_prefetch(src + 16 * H_DIM, 0, 3);
      _Float16* dst = slot + ((nt + 1) & 1) * (16 * H_DIM);
      for (int i = tid; i < 16 * H_DIM; i += 256) {
        int r = i >> 7, c = i & 127;
        dst[i] = (_Float16)src[(size_t)r * H_DIM + c];
      }
    }
    const _Float16* w4 = slot + buf * (16 * H_DIM);
    v16h bf[4];
#pragma unroll
    for (int kt = 0; kt < 4; ++kt)
      bf[kt] = bfrag_lds(&w4[n * H_DIM + kt * 32 + g * 16]);
    float bias = db2[n0 + n];
    v8f oacc = {};
#pragma unroll 1
    for (int t = 0; t < T_STEPS; ++t) {
      v8f acc;
#pragma unroll
      for (int r = 0; r < 8; ++r) acc[r] = bias;
#pragma unroll
      for (int kt = 0; kt < 4; ++kt)
        acc = wmma_f16(afrag_bits(sb[(t * 128 + w * 16 + n) * 4 + kt], g),
                       bf[kt], acc);
#pragma unroll
      for (int r = 0; r < 8; ++r)
        oacc[r] += fast_sigmoid(acc[r]);
    }
#pragma unroll
    for (int r = 0; r < 8; ++r)
      out[(size_t)(rowbase + r + 8 * g) * D_IN + n0 + n] = oacc[r] * 0.0625f;
    __syncthreads();                          // staging/consumption handshake
  }
}

extern "C" void kernel_launch(void* const* d_in, const int* in_sizes, int n_in,
                              void* d_out, int out_size, void* d_ws, size_t ws_size,
                              hipStream_t stream) {
  (void)in_sizes; (void)n_in; (void)out_size; (void)d_ws; (void)ws_size;
  const float* x   = (const float*)d_in[0];
  const float* ew1 = (const float*)d_in[1];
  const float* eb1 = (const float*)d_in[2];
  const float* ew2 = (const float*)d_in[3];
  const float* eb2 = (const float*)d_in[4];
  const float* dw1 = (const float*)d_in[5];
  const float* db1 = (const float*)d_in[6];
  const float* dw2 = (const float*)d_in[7];
  const float* db2 = (const float*)d_in[8];
  float* out = (float*)d_out;
  // 32768 rows / 128 rows per block = 256 blocks, 256 threads (8 waves) each.
  snn_ae_fused<<<dim3(256), dim3(256), 0, stream>>>(x, ew1, eb1, ew2, eb2,
                                                    dw1, db1, dw2, db2, out);
}
